// ScaledDotProductAttention_83597243450155
// MI455X (gfx1250) — matrix operand order repaired and checked by reading, not compiled, not run on hardware
//
#include <hip/hip_runtime.h>

#define BATCH_H 24      // B*H
#define SEQ     4096
#define HEADD   64
#define BM      128     // query rows per workgroup (8 waves x 16)
#define BN      64      // kv rows per block
#define NWAVES  8
#define LSTR    66      // LDS row stride in bf16 (even, odd dword count)

typedef __bf16 bf16;
typedef __attribute__((ext_vector_type(16))) __bf16 v16bf;
typedef __attribute__((ext_vector_type(8)))  float  v8f;

union AF { v16bf v; bf16 e[16]; unsigned int u[8]; };
union CF { v8f  v; float f[8]; };
union PK { bf16 h[2]; unsigned int u; };

// exchange with lane^16 (identity position swap between the two 16-lane rows)
__device__ __forceinline__ int xhalf_i(int x) {
    return __builtin_amdgcn_permlanex16(x, x, 0x76543210, (int)0xFEDCBA98u, false, false);
}
__device__ __forceinline__ float xhalf_f(float x) {
    return __int_as_float(xhalf_i(__float_as_int(x)));
}

__global__ __launch_bounds__(256)
void fa_fwd_wmma_bf16(const float* __restrict__ Q, const float* __restrict__ K,
                      const float* __restrict__ V, float* __restrict__ O) {
    __shared__ __attribute__((aligned(16))) bf16 sK [BN][LSTR];      // [kv][d]
    __shared__ __attribute__((aligned(16))) bf16 sVt[HEADD][LSTR];   // [d][kv]

    const int tid  = threadIdx.x;
    const int w    = tid >> 5;
    const int lane = tid & 31;
    const int half = lane >> 4;
    const int l16  = lane & 15;

    const int bh  = blockIdx.x % BATCH_H;
    const int qt  = blockIdx.x / BATCH_H;
    const int q0w = qt * BM + w * 16;
    const int nblk = 2 * qt + 2;
    const int my_last = (q0w + 15) >> 6;
    const int qg = q0w + l16;                    // this lane's query row

    const float SCL = 0.18033688011112042f;      // log2(e)/sqrt(64), folded into Q

    // ---- Q^T as two B fragments (32x16: K=d chunk, N=q), pre-scaled ----
    AF bQ[2];
    {
        const float* qrow = Q + ((size_t)bh * SEQ + qg) * HEADD;
#pragma unroll
        for (int c = 0; c < 2; ++c)
#pragma unroll
            for (int j = 0; j < 8; ++j) {
                int kb = c * 32 + half * 16 + 2 * j;
                float2 f = *(const float2*)(qrow + kb);
                bQ[c].e[2 * j]     = (bf16)(f.x * SCL);
                bQ[c].e[2 * j + 1] = (bf16)(f.y * SCL);
            }
    }

    // per-lane scalar softmax state (lane <-> one query row); O^T accumulators
    float row_max = -3.0e38f, row_sum = 0.0f;
    CF acc[4];
#pragma unroll
    for (int d = 0; d < 4; ++d)
#pragma unroll
        for (int i = 0; i < 8; ++i) acc[d].f[i] = 0.0f;

    for (int bk = 0; bk < nblk; ++bk) {
        const int kv0 = bk * BN;

        __syncthreads();
        // ---- stage K (row major) and V (transposed) as bf16; prefetch next ----
        {
            int r  = tid >> 2;           // 0..63 kv row
            int c0 = (tid & 3) * 16;     // 0,16,32,48
            const float* kbase = K + ((size_t)bh * SEQ + kv0 + r) * HEADD + c0;
            const float* vbase = V + ((size_t)bh * SEQ + kv0 + r) * HEADD + c0;
            if (bk + 1 < nblk) {
                __builtin_prefetch(kbase + BN * HEADD, 0, 3);
                __builtin_prefetch(vbase + BN * HEADD, 0, 3);
            }
            union { bf16 e[16]; unsigned int u[8]; } kb;
            float4 f0 = ((const float4*)kbase)[0], f1 = ((const float4*)kbase)[1];
            float4 f2 = ((const float4*)kbase)[2], f3 = ((const float4*)kbase)[3];
            kb.e[0]=(bf16)f0.x; kb.e[1]=(bf16)f0.y; kb.e[2]=(bf16)f0.z; kb.e[3]=(bf16)f0.w;
            kb.e[4]=(bf16)f1.x; kb.e[5]=(bf16)f1.y; kb.e[6]=(bf16)f1.z; kb.e[7]=(bf16)f1.w;
            kb.e[8]=(bf16)f2.x; kb.e[9]=(bf16)f2.y; kb.e[10]=(bf16)f2.z; kb.e[11]=(bf16)f2.w;
            kb.e[12]=(bf16)f3.x; kb.e[13]=(bf16)f3.y; kb.e[14]=(bf16)f3.z; kb.e[15]=(bf16)f3.w;
            unsigned int* dst = (unsigned int*)&sK[r][c0];
#pragma unroll
            for (int j = 0; j < 8; ++j) dst[j] = kb.u[j];

            float4 g0 = ((const float4*)vbase)[0], g1 = ((const float4*)vbase)[1];
            float4 g2 = ((const float4*)vbase)[2], g3 = ((const float4*)vbase)[3];
            sVt[c0+ 0][r]=(bf16)g0.x; sVt[c0+ 1][r]=(bf16)g0.y; sVt[c0+ 2][r]=(bf16)g0.z; sVt[c0+ 3][r]=(bf16)g0.w;
            sVt[c0+ 4][r]=(bf16)g1.x; sVt[c0+ 5][r]=(bf16)g1.y; sVt[c0+ 6][r]=(bf16)g1.z; sVt[c0+ 7][r]=(bf16)g1.w;
            sVt[c0+ 8][r]=(bf16)g2.x; sVt[c0+ 9][r]=(bf16)g2.y; sVt[c0+10][r]=(bf16)g2.z; sVt[c0+11][r]=(bf16)g2.w;
            sVt[c0+12][r]=(bf16)g3.x; sVt[c0+13][r]=(bf16)g3.y; sVt[c0+14][r]=(bf16)g3.z; sVt[c0+15][r]=(bf16)g3.w;
        }
        __syncthreads();

        if (bk > my_last) continue;

        // ---- S^T = K (Q*scl)^T : four 16x16 tiles, lane = q, VGPR = kv ----
        CF sA[4];
#pragma unroll
        for (int t = 0; t < 4; ++t) {
#pragma unroll
            for (int i = 0; i < 8; ++i) sA[t].f[i] = 0.0f;
#pragma unroll
            for (int c = 0; c < 2; ++c) {
                AF a;                       // A layout: lane = kv row, K contiguous per half
                int r = t * 16 + l16;
#pragma unroll
                for (int j = 0; j < 8; ++j) {
                    int kb = c * 32 + half * 16 + 2 * j;
                    a.u[j] = *(const unsigned int*)&sK[r][kb];
                }
                sA[t].v = __builtin_amdgcn_wmma_f32_16x16x32_bf16(
                    false, a.v, false, bQ[c].v, (short)0, sA[t].v, false, false);
            }
        }

        // ---- causal mask: only diagonal blocks (wave-uniform) ----
        if (kv0 + (BN - 1) > q0w) {
#pragma unroll
            for (int t = 0; t < 4; ++t)
#pragma unroll
                for (int i = 0; i < 8; ++i) {
                    int ng = kv0 + t * 16 + half * 8 + i;
                    if (ng > qg) sA[t].f[i] = -3.0e38f;
                }
        }

        // ---- row max: in-lane over 32 regs, then one half-exchange ----
        float mm[8];
#pragma unroll
        for (int i = 0; i < 8; ++i)
            mm[i] = fmaxf(fmaxf(sA[0].f[i], sA[1].f[i]),
                          fmaxf(sA[2].f[i], sA[3].f[i]));
        float rmax = mm[0];
#pragma unroll
        for (int i = 1; i < 8; ++i) rmax = fmaxf(rmax, mm[i]);
        rmax = fmaxf(rmax, xhalf_f(rmax));

        // ---- rescale only when the max moved (per-lane scalar corr) ----
        if (__any(rmax > row_max)) {
            float nm   = fmaxf(row_max, rmax);
            float corr = __builtin_amdgcn_exp2f(row_max - nm);
            row_max = nm;
            row_sum *= corr;
#pragma unroll
            for (int d = 0; d < 4; ++d)
#pragma unroll
                for (int i = 0; i < 8; ++i) acc[d].f[i] *= corr;
        }

        // ---- P = exp2(S^T - max): pack bf16 pairs in-register, sum in-lane ----
        unsigned int pkv[4][4];
        float psum = 0.0f;
#pragma unroll
        for (int t = 0; t < 4; ++t)
#pragma unroll
            for (int j = 0; j < 4; ++j) {
                float p0 = __builtin_amdgcn_exp2f(sA[t].f[2 * j]     - row_max);
                float p1 = __builtin_amdgcn_exp2f(sA[t].f[2 * j + 1] - row_max);
                psum += p0 + p1;
                PK pp; pp.h[0] = (bf16)p0; pp.h[1] = (bf16)p1;
                pkv[t][j] = pp.u;
            }
        psum += xhalf_f(psum);
        row_sum += psum;

        // ---- build P^T B-fragments: one half-swap per 4 dwords ----
        AF bP[2];
#pragma unroll
        for (int c = 0; c < 2; ++c)
#pragma unroll
            for (int j = 0; j < 4; ++j) {
                unsigned int send = half ? pkv[2 * c][j] : pkv[2 * c + 1][j];
                unsigned int recv = (unsigned int)xhalf_i((int)send);
                bP[c].u[j]     = half ? recv : pkv[2 * c][j];
                bP[c].u[j + 4] = half ? pkv[2 * c + 1][j] : recv;
            }

        // ---- O^T += V^T P^T : 8 wmma ----
#pragma unroll
        for (int t = 0; t < 4; ++t)
#pragma unroll
            for (int c = 0; c < 2; ++c) {
                AF a;                       // A layout: lane = d row
                int r = t * 16 + l16;
#pragma unroll
                for (int j = 0; j < 8; ++j) {
                    int kb = c * 32 + half * 16 + 2 * j;
                    a.u[j] = *(const unsigned int*)&sVt[r][kb];
                }
                acc[t].v = __builtin_amdgcn_wmma_f32_16x16x32_bf16(
                    false, a.v, false, bP[c].v, (short)0, acc[t].v, false, false);
            }
    }

    // ---- finalize: one rcp per lane, vectorized 128-bit stores ----
    float inv = 1.0f / row_sum;
    float* obase = O + ((size_t)bh * SEQ + qg) * HEADD + half * 8;
#pragma unroll
    for (int t = 0; t < 4; ++t) {
        float4 o0 = make_float4(acc[t].f[0] * inv, acc[t].f[1] * inv,
                                acc[t].f[2] * inv, acc[t].f[3] * inv);
        float4 o1 = make_float4(acc[t].f[4] * inv, acc[t].f[5] * inv,
                                acc[t].f[6] * inv, acc[t].f[7] * inv);
        ((float4*)(obase + t * 16))[0] = o0;
        ((float4*)(obase + t * 16))[1] = o1;
    }
}

extern "C" void kernel_launch(void* const* d_in, const int* in_sizes, int n_in,
                              void* d_out, int out_size, void* d_ws, size_t ws_size,
                              hipStream_t stream) {
    const float* q = (const float*)d_in[0];
    const float* k = (const float*)d_in[1];
    const float* v = (const float*)d_in[2];
    float* o = (float*)d_out;
    dim3 grid(BATCH_H * (SEQ / BM));   // 768 workgroups, 256 threads (8 waves)
    fa_fwd_wmma_bf16<<<grid, 256, 0, stream>>>(q, k, v, o);
}
